// BSIQuantizedLinear_81432579932605
// MI455X (gfx1250) — matrix operand order repaired and checked by reading, not compiled, not run on hardware
//
#include <hip/hip_runtime.h>
#include <hip/hip_bf16.h>

typedef _Float16 v16h __attribute__((ext_vector_type(16)));
typedef _Float16 v8h  __attribute__((ext_vector_type(8)));
typedef float    v8f  __attribute__((ext_vector_type(8)));
typedef int      i32x4 __attribute__((vector_size(16)));

#define AS_GLOBAL __attribute__((address_space(1)))
#define AS_SHARED __attribute__((address_space(3)))

#ifndef __has_builtin
#define __has_builtin(x) 0
#endif

#if __has_builtin(__builtin_amdgcn_global_load_async_to_lds_b128)
#define HAVE_ASYNC_COPY 1
#else
#define HAVE_ASYNC_COPY 0
#endif

// Problem size (fixed by the reference): out[B, D_OUT] = qx[B,K] * qW[D_OUT,K]^T + b
constexpr int MDIM = 4096;   // B (rows of x)
constexpr int NDIM = 4096;   // D_OUT (rows of W)
constexpr int KDIM = 4096;   // D_IN
constexpr int BM = 128;      // block tile M
constexpr int BN = 128;      // block tile N
constexpr int KC = 64;       // K-step staged in LDS
constexpr int LDA = KC + 8;  // LDS row stride in halves (+16B pad vs bank conflicts)

__device__ __forceinline__ void wait_async0() {
#if __has_builtin(__builtin_amdgcn_s_wait_asynccnt)
  __builtin_amdgcn_s_wait_asynccnt(0);
#else
  asm volatile("s_wait_asynccnt 0" ::: "memory");
#endif
}

#if HAVE_ASYNC_COPY
// 16B direct global->LDS DMA (ASYNCcnt-tracked), bypasses VGPRs.
// Builtin signature (from hipcc diagnostic): (i32x4 addrspace(1)*, i32x4 addrspace(3)*, imm offset, imm cpol)
__device__ __forceinline__ void async_copy16(const _Float16* g, _Float16* l) {
  __builtin_amdgcn_global_load_async_to_lds_b128(
      (AS_GLOBAL i32x4*)(g), (AS_SHARED i32x4*)(l), 0, 0);
}
#endif

// Quantize 8 fp32 values to scaled-integer f16: h = (f16)rint(100*v)  (exact: |v*100| < 2048)
__device__ __forceinline__ v8h quant8(float4 a, float4 b) {
  v8h o;
  o[0] = (_Float16)__builtin_rintf(a.x * 100.0f);
  o[1] = (_Float16)__builtin_rintf(a.y * 100.0f);
  o[2] = (_Float16)__builtin_rintf(a.z * 100.0f);
  o[3] = (_Float16)__builtin_rintf(a.w * 100.0f);
  o[4] = (_Float16)__builtin_rintf(b.x * 100.0f);
  o[5] = (_Float16)__builtin_rintf(b.y * 100.0f);
  o[6] = (_Float16)__builtin_rintf(b.z * 100.0f);
  o[7] = (_Float16)__builtin_rintf(b.w * 100.0f);
  return o;
}

// Pre-pass: fp32 -> scaled-integer f16 (8 elements / thread, 16B stores)
__global__ __launch_bounds__(256) void bsi_quant_kernel(const float* __restrict__ in,
                                                        _Float16* __restrict__ out,
                                                        int n8) {
  int i = blockIdx.x * blockDim.x + threadIdx.x;
  if (i >= n8) return;
  const float4* p = reinterpret_cast<const float4*>(in) + (size_t)i * 2;
  float4 a = p[0];
  float4 b = p[1];
  reinterpret_cast<v8h*>(out)[i] = quant8(a, b);
}

// Double-buffered, async-staged f16 WMMA GEMM:
//   out[m,n] = (sum_k Aq[m,k]*Wq[n,k]) * 1e-4 + bias[n]
// FUSED=true: quantize fp32 operands during (synchronous) LDS staging.
template <bool FUSED>
__global__ __launch_bounds__(256) void bsi_gemm_kernel(const _Float16* __restrict__ Aq,
                                                       const _Float16* __restrict__ Wq,
                                                       const float* __restrict__ Af,
                                                       const float* __restrict__ Wf,
                                                       const float* __restrict__ bias,
                                                       float* __restrict__ out) {
  __shared__ __align__(16) _Float16 sA[2][BM * LDA];
  __shared__ __align__(16) _Float16 sB[2][BN * LDA];

  const int tid  = threadIdx.x;
  const int lane = tid & 31;
  const int wave = tid >> 5;
  const int wm   = wave & 3;   // 4 waves along M  -> wave tile 32 rows
  const int wn   = wave >> 2;  // 2 waves along N  -> wave tile 64 cols
  const int mRow = lane & 15;  // row/col within 16x16 tile
  const int hsel = lane >> 4;  // half-wave select

  const int blockM = blockIdx.y * BM;
  const int blockN = blockIdx.x * BN;

  // Staging coords: chunk c = tid + i*256 -> row = tid/8 + 32*i, ko = (tid%8)*8 halves
  const int row0 = tid >> 3;
  const int ko   = (tid & 7) << 3;

  // Stage one 128x64 f16 tile of A and of W into LDS buffer `buf` for K-offset k0.
  auto stage = [&](int buf, int k0) {
    _Float16* la = &sA[buf][0];
    _Float16* lb = &sB[buf][0];
#pragma unroll
    for (int i = 0; i < 4; ++i) {
      int row  = row0 + i * 32;
      int loff = row * LDA + ko;
      if constexpr (FUSED) {
        const float* pa = &Af[(size_t)(blockM + row) * KDIM + k0 + ko];
        const float* pw = &Wf[(size_t)(blockN + row) * KDIM + k0 + ko];
        float4 a0 = *reinterpret_cast<const float4*>(pa);
        float4 a1 = *reinterpret_cast<const float4*>(pa + 4);
        float4 w0 = *reinterpret_cast<const float4*>(pw);
        float4 w1 = *reinterpret_cast<const float4*>(pw + 4);
        *reinterpret_cast<v8h*>(la + loff) = quant8(a0, a1);
        *reinterpret_cast<v8h*>(lb + loff) = quant8(w0, w1);
      } else {
#if HAVE_ASYNC_COPY
        async_copy16(&Aq[(size_t)(blockM + row) * KDIM + k0 + ko], la + loff);
        async_copy16(&Wq[(size_t)(blockN + row) * KDIM + k0 + ko], lb + loff);
#else
        *reinterpret_cast<v8h*>(la + loff) =
            *reinterpret_cast<const v8h*>(&Aq[(size_t)(blockM + row) * KDIM + k0 + ko]);
        *reinterpret_cast<v8h*>(lb + loff) =
            *reinterpret_cast<const v8h*>(&Wq[(size_t)(blockN + row) * KDIM + k0 + ko]);
#endif
      }
    }
  };

  v8f acc[2][4] = {};

  constexpr int nT = KDIM / KC;
  stage(0, 0);  // prologue: tile 0 -> buf 0

  for (int t = 0; t < nT; ++t) {
    const int cur = t & 1;
    if constexpr (!FUSED) {
#if HAVE_ASYNC_COPY
      wait_async0();  // this wave's DMA for tile t has landed in LDS
#endif
    }
    __syncthreads();  // all waves' tile-t data visible; all done reading buf[cur^1]

    if (t + 1 < nT) stage(cur ^ 1, (t + 1) * KC);  // overlap DMA with WMMA below

    const _Float16* a  = sA[cur];
    const _Float16* bb = sB[cur];
#pragma unroll
    for (int kk = 0; kk < KC; kk += 32) {
      v16h af[2];
      v16h bf[4];
      // A fragment (16x32): elems 0-7 <- K = kk + hsel*8 + i ; elems 8-15 <- +16
#pragma unroll
      for (int mt = 0; mt < 2; ++mt) {
        int r = wm * 32 + mt * 16 + mRow;
        v8h* h = reinterpret_cast<v8h*>(&af[mt]);
        h[0] = *reinterpret_cast<const v8h*>(&a[r * LDA + kk + hsel * 8]);
        h[1] = *reinterpret_cast<const v8h*>(&a[r * LDA + kk + 16 + hsel * 8]);
      }
      // B fragment (32x16): lane holds column (lane%16); elems 0-7 <- K = kk + hsel*8 + i ; elems 8-15 <- +16
#pragma unroll
      for (int nt = 0; nt < 4; ++nt) {
        int r = wn * 64 + nt * 16 + mRow;  // W row == output column
        v8h* h = reinterpret_cast<v8h*>(&bf[nt]);
        h[0] = *reinterpret_cast<const v8h*>(&bb[r * LDA + kk + hsel * 8]);
        h[1] = *reinterpret_cast<const v8h*>(&bb[r * LDA + kk + 16 + hsel * 8]);
      }
#pragma unroll
      for (int mt = 0; mt < 2; ++mt)
#pragma unroll
        for (int nt = 0; nt < 4; ++nt)
          acc[mt][nt] = __builtin_amdgcn_wmma_f32_16x16x32_f16(
              false, af[mt], false, bf[nt], (short)0, acc[mt][nt], false, false);
    }
  }

  // Epilogue: out = acc * 1e-4 + bias   (D layout: row r + 8*hsel, col lane%16)
  const float inv = 1.0f / (100.0f * 100.0f);
#pragma unroll
  for (int mt = 0; mt < 2; ++mt) {
#pragma unroll
    for (int nt = 0; nt < 4; ++nt) {
      int col  = blockN + wn * 64 + nt * 16 + mRow;
      float bv = bias[col];
#pragma unroll
      for (int r = 0; r < 8; ++r) {
        int row = blockM + wm * 32 + mt * 16 + r + hsel * 8;
        out[(size_t)row * NDIM + col] = acc[mt][nt][r] * inv + bv;
      }
    }
  }
}

extern "C" void kernel_launch(void* const* d_in, const int* in_sizes, int n_in,
                              void* d_out, int out_size, void* d_ws, size_t ws_size,
                              hipStream_t stream) {
  (void)in_sizes; (void)n_in; (void)out_size;
  const float* x    = (const float*)d_in[0];
  const float* W    = (const float*)d_in[1];
  const float* bias = (const float*)d_in[2];
  float* out        = (float*)d_out;

  const size_t elems = (size_t)MDIM * KDIM;
  const size_t need  = 2 * elems * sizeof(_Float16);  // 64 MiB
  dim3 grid(NDIM / BN, MDIM / BM);

  if (ws_size >= need) {
    _Float16* xq = (_Float16*)d_ws;
    _Float16* wq = xq + elems;
    int n8 = (int)(elems / 8);
    bsi_quant_kernel<<<n8 / 256, 256, 0, stream>>>(x, xq, n8);
    bsi_quant_kernel<<<n8 / 256, 256, 0, stream>>>(W, wq, n8);
    bsi_gemm_kernel<false><<<grid, 256, 0, stream>>>(xq, wq, nullptr, nullptr, bias, out);
  } else {
    bsi_gemm_kernel<true><<<grid, 256, 0, stream>>>(nullptr, nullptr, x, W, bias, out);
  }
}
